// RNet1_47639777247625
// MI455X (gfx1250) — matrix operand order repaired and checked by reading, not compiled, not run on hardware
//
#include <hip/hip_runtime.h>
#include <hip/hip_bf16.h>

#define HDIM 64
#define WROW 72   // padded LDS row stride in halfs (144B) -> conflict-free rows

typedef __attribute__((ext_vector_type(16))) _Float16 v16h;
typedef __attribute__((ext_vector_type(8)))  _Float16 v8h;
typedef __attribute__((ext_vector_type(8)))  float    v8f;

static __device__ __forceinline__ float sigmoidf_(float x){
  return 1.0f / (1.0f + __expf(-x));
}
static __device__ __forceinline__ float eluf_(float x){
  return x > 0.0f ? x : (__expf(x) - 1.0f);
}

union V8HI { v8h h; int i[4]; };

// swap data between lane l and lane l^16 (half-wave exchange)
static __device__ __forceinline__ v8h shfl_xor16(v8h x){
  V8HI u, r; u.h = x;
#pragma unroll
  for (int i = 0; i < 4; ++i) r.i[i] = __shfl_xor(u.i[i], 16, 32);
  return r.h;
}

static __device__ __forceinline__ v16h vcat(v8h lo, v8h hi){
  v16h r;
#pragma unroll
  for (int i = 0; i < 8; ++i){ r[i] = lo[i]; r[i + 8] = hi[i]; }
  return r;
}

static __device__ __forceinline__ v8f wmma_acc(v16h a, v16h b, v8f c){
  // D = A(16x32 f16) * B(32x16 f16) + C(16x16 f32)
  return __builtin_amdgcn_wmma_f32_16x16x32_f16(false, a, false, b, (short)0, c,
                                                false, false);
}

// Cooperatively convert a 64x64 f32 weight matrix to f16 in LDS
// (row-major, padded row stride WROW). 256 threads, 16 elems each,
// stored as two aligned 16B vectors -> ds_store_b128.
static __device__ __forceinline__ void stage_w(const float* __restrict__ W,
                                               _Float16* __restrict__ dst,
                                               int tid){
  const int row = tid >> 2;            // 0..63
  const int c0  = (tid & 3) << 4;      // 0,16,32,48
  const float* src = W + row * HDIM + c0;
  _Float16*   d   = dst + row * WROW + c0;
  v8h a, b;
#pragma unroll
  for (int i = 0; i < 8; ++i){
    a[i] = (_Float16)src[i];
    b[i] = (_Float16)src[8 + i];
  }
  *(v8h*)(d)     = a;
  *(v8h*)(d + 8) = b;
}

// Read the 4x2 A-layout fragments of the staged 64x64 f16 matrix.
// A tile jt covers rows j = jt*16 + (lane&15); A operand, this file's contiguous K order:
//   lane<16 : halfs = K { kb..kb+15 }, kb = 0
//   lane>=16: same with kb = 16
// Each 8-half group is a 16B-aligned ds_load_b128.
static __device__ __forceinline__ void load_frags_lds(const _Float16* __restrict__ src,
                                                      int lane, v16h f[4][2]){
  const int m  = lane & 15;
  const int kb = (lane >> 4) << 4;
#pragma unroll
  for (int jt = 0; jt < 4; ++jt){
    const _Float16* r = src + (jt * 16 + m) * WROW;
#pragma unroll
    for (int kt = 0; kt < 2; ++kt){
      v8h lo = *(const v8h*)(r + kt * 32 + kb);
      v8h hi = *(const v8h*)(r + kt * 32 + kb + 8);
      f[jt][kt] = vcat(lo, hi);
    }
  }
}

__global__ __launch_bounds__(256, 4) void rnet_fused(
    const float* __restrict__ sxy0, const float* __restrict__ sxy,
    const float* __restrict__ oxy,
    const float* __restrict__ W_h0, const float* __restrict__ b_h0,
    const float* __restrict__ W_h1, const float* __restrict__ b_h1,
    const float* __restrict__ W_ih, const float* __restrict__ b_ih,
    const float* __restrict__ W_hh, const float* __restrict__ b_hh,
    const float* __restrict__ W_psi0, const float* __restrict__ b_psi0,
    const float* __restrict__ W_psi, const float* __restrict__ b_psi,
    const float* __restrict__ W_p, const float* __restrict__ b_p,
    float* __restrict__ out_psi0, float* __restrict__ out_psi,
    float* __restrict__ out_p, int N, int T)
{
  // f16 weight buffer (W_h1 first, then W_hh) + broadcast tables:
  //   ldsIB[j] = { W_ih[j], b_ih[j] + b_hh[j] }   (RNN step C-init)
  //   ldsW [j] = { W_psi[j], W_p[j] }             (t >= 1 readout)
  //   ldsW0[j] = { W_psi0[j], W_p[j] }            (t = 0 readout)
  __shared__ _Float16 ldsWf[HDIM * WROW];
  __shared__ float2 ldsIB[HDIM];
  __shared__ float2 ldsW [HDIM];
  __shared__ float2 ldsW0[HDIM];

  const int tid = threadIdx.x;
  stage_w(W_h1, ldsWf, tid);
  if (tid < HDIM){
    ldsIB[tid] = make_float2(W_ih[tid], b_ih[tid] + b_hh[tid]);
    ldsW [tid] = make_float2(W_psi [tid], W_p[tid]);
    ldsW0[tid] = make_float2(W_psi0[tid], W_p[tid]);
  }
  __syncthreads();

  const int  lane  = tid & 31;
  const int  wave  = tid >> 5;
  const long site0 = ((long)blockIdx.x * 8 + wave) * 16;

  const int  hi = lane >> 4;              // half-wave id
  const int  m  = lane & 15;              // site within tile / matrix column N
  long site = site0 + m;
  const bool wr = (hi == 0) && (site < (long)N);
  if (site >= (long)N) site = (long)N - 1;   // clamp: tail waves run masked

  const float w_ov  = W_p[HDIM];          // weight multiplying oxy
  const float bp    = b_p[0];
  const float bpsi  = b_psi[0];
  const float bpsi0 = b_psi0[0];
  const int   Tm1   = T - 1;
  const int   jbase = hi * 8;             // this lane's feature offset in a tile

  // ---- hs = elu(x0*W_h0 + b_h0), built directly in B-layout (hsT) ----
  // B operand, this file's contiguous K order: lane n = lane&15, halfs = K hi*16 + 0..15
  const float x0   = sxy0[site];
  const int   kb16 = hi << 4;
  v16h hfrag0, hfrag1;
#pragma unroll
  for (int i = 0; i < 16; ++i){
    const int k0 = kb16 + i;
    const int k1 = 32 + kb16 + i;
    hfrag0[i] = (_Float16)eluf_(x0 * W_h0[k0] + b_h0[k0]);
    hfrag1[i] = (_Float16)eluf_(x0 * W_h0[k1] + b_h0[k1]);
  }

  // ---- h0T = elu(W_h1 * hsT + b_h1): 8 WMMAs ----
  v8f acc[4];                              // D-layout: lane = site, rows = j
  {
    v16h wfr[4][2];
    load_frags_lds(ldsWf, lane, wfr);
#pragma unroll
    for (int jt = 0; jt < 4; ++jt){
      v8f c;
#pragma unroll
      for (int r = 0; r < 8; ++r) c[r] = b_h1[jt * 16 + jbase + r];
      c = wmma_acc(wfr[jt][0], hfrag0, c);
      c = wmma_acc(wfr[jt][1], hfrag1, c);
#pragma unroll
      for (int r = 0; r < 8; ++r) c[r] = eluf_(c[r]);
      acc[jt] = c;
    }
  }

  // ---- t = 0 readout: psi0 and p[:,0,:] ----
  {
    float a = 0.f, b = 0.f;
#pragma unroll
    for (int jt = 0; jt < 4; ++jt)
#pragma unroll
      for (int r = 0; r < 8; ++r){
        const float2 w = ldsW0[jt * 16 + jbase + r];
        a = fmaf(acc[jt][r], w.x, a);
        b = fmaf(acc[jt][r], w.y, b);
      }
    a += __shfl_xor(a, 16, 32);            // other half holds remaining j's
    b += __shfl_xor(b, 16, 32);
    if (wr){
      out_psi0[site] = sigmoidf_(a + bpsi0);
#pragma unroll
      for (int j = 0; j < 2; ++j){
        const long oi = site * (2 * (long)T) + j;
        out_p[oi] = sigmoidf_(b + oxy[oi] * w_ov + bp);
      }
    }
  }

  // ---- swap the LDS weight buffer to W_hh ----
  __syncthreads();                 // everyone done reading W_h1 fragments
  stage_w(W_hh, ldsWf, tid);
  __syncthreads();

  // ---- recurrent loop: h <- relu(W_hh*hT + x_t*W_ih + b) ----
  float xt = sxy[site * (long)Tm1];        // x for t = 1, prefetched
  for (int t = 1; t < T; ++t){
    // prefetch next step's x before the dependent WMMA chain
    float xt_next = 0.0f;
    if (t < Tm1) xt_next = sxy[site * (long)Tm1 + t];

    // weight fragments from LDS (compiler may hoist; cheap either way)
    v16h wfr[4][2];
    load_frags_lds(ldsWf, lane, wfr);

    // convert D-layout acc (lane=site, rows=j) into B-layout hT fragments:
    // pack own 8 rows to f16, exchange with partner lane (l^16) which holds
    // the other 8 rows of each 16-row group, then concatenate by K order.
    v8h hh[4], ph[4];
#pragma unroll
    for (int jt = 0; jt < 4; ++jt){
      v8h g;
#pragma unroll
      for (int i = 0; i < 8; ++i) g[i] = (_Float16)acc[jt][i];
      hh[jt] = g;
      ph[jt] = shfl_xor16(g);
    }
    const v16h b0 = hi ? vcat(ph[1], hh[1]) : vcat(hh[0], ph[0]); // K  0..31
    const v16h b1 = hi ? vcat(ph[3], hh[3]) : vcat(hh[2], ph[2]); // K 32..63

#pragma unroll
    for (int jt = 0; jt < 4; ++jt){
      v8f c;
#pragma unroll
      for (int r = 0; r < 8; ++r){
        const float2 ib = ldsIB[jt * 16 + jbase + r];
        c[r] = fmaf(xt, ib.x, ib.y);
      }
      c = wmma_acc(wfr[jt][0], b0, c);
      c = wmma_acc(wfr[jt][1], b1, c);
#pragma unroll
      for (int r = 0; r < 8; ++r) c[r] = fmaxf(c[r], 0.0f);
      acc[jt] = c;
    }

    // readout: psi[:, t-1] and p[:, t, :]
    float a = 0.f, b = 0.f;
#pragma unroll
    for (int jt = 0; jt < 4; ++jt)
#pragma unroll
      for (int r = 0; r < 8; ++r){
        const float2 w = ldsW[jt * 16 + jbase + r];
        a = fmaf(acc[jt][r], w.x, a);
        b = fmaf(acc[jt][r], w.y, b);
      }
    a += __shfl_xor(a, 16, 32);
    b += __shfl_xor(b, 16, 32);
    if (wr){
      out_psi[site * (long)Tm1 + (t - 1)] = sigmoidf_(a + bpsi);
#pragma unroll
      for (int j = 0; j < 2; ++j){
        const long oi = site * (2 * (long)T) + 2 * t + j;
        out_p[oi] = sigmoidf_(b + oxy[oi] * w_ov + bp);
      }
    }
    xt = xt_next;
  }
}

extern "C" void kernel_launch(void* const* d_in, const int* in_sizes, int n_in,
                              void* d_out, int out_size, void* d_ws, size_t ws_size,
                              hipStream_t stream) {
  const float* sxy0   = (const float*)d_in[0];
  const float* sxy    = (const float*)d_in[1];
  const float* oxy    = (const float*)d_in[2];
  const float* W_h0   = (const float*)d_in[3];
  const float* b_h0   = (const float*)d_in[4];
  const float* W_h1   = (const float*)d_in[5];
  const float* b_h1   = (const float*)d_in[6];
  const float* W_ih   = (const float*)d_in[7];
  const float* b_ih   = (const float*)d_in[8];
  const float* W_hh   = (const float*)d_in[9];
  const float* b_hh   = (const float*)d_in[10];
  const float* W_psi0 = (const float*)d_in[11];
  const float* b_psi0 = (const float*)d_in[12];
  const float* W_psi  = (const float*)d_in[13];
  const float* b_psi  = (const float*)d_in[14];
  const float* W_p    = (const float*)d_in[15];
  const float* b_p    = (const float*)d_in[16];

  const int N = in_sizes[0];            // sxy0 is [1, N, 1]
  const int T = in_sizes[1] / N + 1;    // sxy is [N, T-1, 1] -> T = 20

  float* out      = (float*)d_out;
  float* out_psi0 = out;                       // N
  float* out_psi  = out + (size_t)N;           // N*(T-1)
  float* out_p    = out + (size_t)N * T;       // N*T*2

  const int sites_per_block = 8 * 16;          // 8 waves x 16-site tiles
  const int blocks = (N + sites_per_block - 1) / sites_per_block;

  rnet_fused<<<blocks, 256, 0, stream>>>(
      sxy0, sxy, oxy, W_h0, b_h0, W_h1, b_h1, W_ih, b_ih, W_hh, b_hh,
      W_psi0, b_psi0, W_psi, b_psi, W_p, b_p,
      out_psi0, out_psi, out_p, N, T);
}
